// UVLearner_65111704207926
// MI455X (gfx1250) — matrix operand order repaired and checked by reading, not compiled, not run on hardware
//
#include <hip/hip_runtime.h>

typedef _Float16 half_t;
typedef __attribute__((ext_vector_type(16))) _Float16 v16h;
typedef __attribute__((ext_vector_type(8)))  _Float16 v8h;
typedef __attribute__((ext_vector_type(8)))  float    v8f;

#define NB    4      // batches
#define CDIM  128    // channels
#define QDIM  4096   // query pixels (64*64)
#define KDIM  4096   // key pixels   (64*64)

// c^-0.5 * log2(e): folded into X during the f16 pre-pass so WMMA logits are
// already in the log2 domain (exp2f == native v_exp_f32, no per-tile muls).
#define SCALE_LOG2E (0.08838834764831845f * 1.4426950408889634f)

// ---------------------------------------------------------------------------
// Pre-pass: transpose [b][c][pix] f32 -> [b][pix][c] f16 for both X and Y.
// X is pre-scaled by SCALE_LOG2E (pure rescale; f16 relative precision
// unaffected). Makes WMMA A/B fragment loads contiguous per lane.
// ---------------------------------------------------------------------------
__global__ __launch_bounds__(256)
void transpose_cvt_f16(const float* __restrict__ x,
                       const float* __restrict__ y,
                       half_t* __restrict__ xt,
                       half_t* __restrict__ yt)
{
    __shared__ float tile[32][33];
    const int which = blockIdx.z >> 2;          // 0 = x, 1 = y
    const int b     = blockIdx.z & 3;
    const float* __restrict__ src = which ? y : x;
    half_t*      __restrict__ dst = which ? yt : xt;
    const float  mul = which ? 1.0f : SCALE_LOG2E;
    const int q0 = blockIdx.x * 32;
    const int c0 = blockIdx.y * 32;
    const int tx = threadIdx.x;                 // 0..31
    const int ty = threadIdx.y;                 // 0..7
#pragma unroll
    for (int j = 0; j < 4; ++j) {
        int c = c0 + ty + j * 8;
        tile[ty + j * 8][tx] = src[((size_t)(b * CDIM + c)) * QDIM + q0 + tx];
    }
    __syncthreads();
#pragma unroll
    for (int j = 0; j < 4; ++j) {
        int q = q0 + ty + j * 8;
        dst[((size_t)(b * QDIM + q)) * CDIM + c0 + tx] =
            (half_t)(tile[tx][ty + j * 8] * mul);
    }
}

// ---------------------------------------------------------------------------
// VALU-pipe 16-lane reductions via DPP row_ror (0x121/0x122/0x124/0x128).
// Rotation-reduce within each 16-lane row leaves the full result in every
// lane and fuses into v_add_f32_dpp / v_max_f32_dpp — no DS-pipe traffic,
// co-executes with XDL WMMA. Control flow is converged wherever these run.
// ---------------------------------------------------------------------------
template <int CTRL>
__device__ __forceinline__ float dpp_rot(float v) {
    return __builtin_bit_cast(float,
        __builtin_amdgcn_update_dpp(0, __builtin_bit_cast(int, v),
                                    CTRL, 0xF, 0xF, true));
}
__device__ __forceinline__ float redsum16(float v) {
    v += dpp_rot<0x121>(v);   // row_ror:1
    v += dpp_rot<0x122>(v);   // row_ror:2
    v += dpp_rot<0x124>(v);   // row_ror:4
    v += dpp_rot<0x128>(v);   // row_ror:8
    return v;
}
__device__ __forceinline__ float redmax16(float v) {
    v = fmaxf(v, dpp_rot<0x121>(v));
    v = fmaxf(v, dpp_rot<0x122>(v));
    v = fmaxf(v, dpp_rot<0x124>(v));
    v = fmaxf(v, dpp_rot<0x128>(v));
    return v;
}

// ---------------------------------------------------------------------------
// Flash-attention style: logits via v_wmma_f32_16x16x32_f16 (f32 accumulate),
// streaming softmax over 4096 keys; the "value" matrix is the fixed grid of
// pixel centers, so PV collapses to per-row reductions (sum p, p*u, p*v).
// One wave per 16-query tile; 4 waves / 128 threads per block (64 queries)
// -> 256 workgroups. B fragments are software-pipelined through an explicit
// double buffer; softmax is amortized over 128-key chunks (8 sub-tiles, two
// image rows -> row-split partial sums give the v-moment with one extra
// reduction instead of doubling all of them).
// ---------------------------------------------------------------------------
__global__ __launch_bounds__(128, 1)
void uv_attn_wmma(const half_t* __restrict__ xt,
                  const half_t* __restrict__ yt,
                  float* __restrict__ out)
{
    const int lane = threadIdx.x & 31;
    const int wave = threadIdx.x >> 5;
    const int b    = blockIdx.y;
    const int q0   = blockIdx.x * 64 + wave * 16;
    const int n    = lane & 15;     // column within 16x16 tile
    const int hi   = lane >> 4;     // half-wave select

    // ---- Load A fragments: 16 queries x 128 channels (4 chunks of K=32).
    // A 16x32 fill (matches B): lanes 0-15 hold K = 0..15,
    // lanes 16-31 hold K = 16..31; M = lane & 15.
    v16h A[4];
    {
        const half_t* abase = xt + ((size_t)(b * QDIM + q0 + n)) * CDIM;
#pragma unroll
        for (int cc = 0; cc < 4; ++cc) {
            A[cc] = *(const v16h*)(abase + cc * 32 + hi * 16);
        }
    }

    float m[8], s[8], su[8], sv[8];
#pragma unroll
    for (int r = 0; r < 8; ++r) {
        m[r] = -__builtin_inff(); s[r] = 0.f; su[r] = 0.f; sv[r] = 0.f;
    }

    // u-coordinate of each key column this lane touches: sub-tile t covers
    // image columns (t&3)*16 + n (chunks are row-aligned), so only 4 values.
    float gu[4];
#pragma unroll
    for (int t = 0; t < 4; ++t) gu[t] = (float)(t * 16 + n) + 0.5f;

    // ---- B fragment double buffer. B 32x16 fill used here (same k order as A): lanes 0-15
    // hold K=0..15 (2/VGPR), lanes 16-31 hold K=16..31; N = lane & 15 ->
    // per-lane 32B contiguous per channel chunk.
    const half_t* ybase = yt + ((size_t)(b * KDIM + n)) * CDIM + hi * 16;
    v16h Bf[2][4];
#pragma unroll
    for (int cc = 0; cc < 4; ++cc)
        Bf[0][cc] = *(const v16h*)(ybase + cc * 32);      // sub-tile 0

    for (int k0 = 0; k0 < KDIM; k0 += 128) {
        v8f acc[8];
        // ---- 32 WMMAs: 8 key sub-tiles x 4 channel chunks.
#pragma unroll
        for (int t = 0; t < 8; ++t) {
            const int cur = t & 1;                        // (k0/16) is even
            // Issue next sub-tile's loads into the other buffer first.
            const int nxt = ((k0 >> 4) + t + 1) & 255;    // wrap at tail
            const half_t* nbase = ybase + (size_t)nxt * (16 * CDIM);
            __builtin_prefetch(nbase + 128 * CDIM, 0, 3); // +1 chunk ahead
#pragma unroll
            for (int cc = 0; cc < 4; ++cc)
                Bf[cur ^ 1][cc] = *(const v16h*)(nbase + cc * 32);
            // 4 chained WMMAs over the channel dimension (K=32 each).
            v8f c = {};
#pragma unroll
            for (int cc = 0; cc < 4; ++cc)
                c = __builtin_amdgcn_wmma_f32_16x16x32_f16(
                        false, A[cc], false, Bf[cur][cc], (short)0, c,
                        false, false);
            acc[t] = c;
        }

        // ---- Online softmax update over this 128-key chunk (log2 domain;
        // scale already folded into A). C/D layout: reg r holds row
        // M = r + hi*8, column N = n. Sub-tiles 0-3 lie on image row
        // (k0>>6), sub-tiles 4-7 on the next row.
        const float gv0 = (float)(k0 >> 6) + 0.5f;
        const float gv1 = gv0 + 1.0f;
#pragma unroll
        for (int r = 0; r < 8; ++r) {
            float l0 = acc[0][r], l1 = acc[1][r], l2 = acc[2][r], l3 = acc[3][r];
            float l4 = acc[4][r], l5 = acc[5][r], l6 = acc[6][r], l7 = acc[7][r];
            float tm = fmaxf(fmaxf(fmaxf(l0, l1), fmaxf(l2, l3)),
                             fmaxf(fmaxf(l4, l5), fmaxf(l6, l7)));
            tm = redmax16(tm);
            float mn = fmaxf(m[r], tm);
            float corr = exp2f(m[r] - mn);
            m[r] = mn;
            float p0 = exp2f(l0 - mn), p1 = exp2f(l1 - mn);
            float p2 = exp2f(l2 - mn), p3 = exp2f(l3 - mn);
            float p4 = exp2f(l4 - mn), p5 = exp2f(l5 - mn);
            float p6 = exp2f(l6 - mn), p7 = exp2f(l7 - mn);
            float pslo = (p0 + p1) + (p2 + p3);           // image row k0>>6
            float pshi = (p4 + p5) + (p6 + p7);           // image row +1
            float ps = redsum16(pslo + pshi);
            float us = redsum16((p0 * gu[0] + p1 * gu[1]) +
                                (p2 * gu[2] + p3 * gu[3]) +
                                (p4 * gu[0] + p5 * gu[1]) +
                                (p6 * gu[2] + p7 * gu[3]));
            float vs = redsum16(gv0 * pslo + gv1 * pshi);
            s[r]  = s[r]  * corr + ps;
            su[r] = su[r] * corr + us;
            sv[r] = sv[r] * corr + vs;
        }
    }

    // ---- Epilogue: lane n==0 of each half holds reduced sums for 8 rows.
    if (n == 0) {
        float2* o = (float2*)out;
#pragma unroll
        for (int r = 0; r < 8; ++r) {
            int q = q0 + hi * 8 + r;
            float inv = 1.0f / s[r];
            // uv / 64 * 2 - 1  ==  uv * 0.03125 - 1
            float uo = su[r] * inv * 0.03125f - 1.0f;
            float vo = sv[r] * inv * 0.03125f - 1.0f;
            o[(size_t)b * QDIM + q] = make_float2(uo, vo);
        }
    }
}

extern "C" void kernel_launch(void* const* d_in, const int* in_sizes, int n_in,
                              void* d_out, int out_size, void* d_ws, size_t ws_size,
                              hipStream_t stream) {
    const float* x = (const float*)d_in[0];
    const float* y = (const float*)d_in[1];
    half_t* xt = (half_t*)d_ws;                         // 4 MiB
    half_t* yt = xt + (size_t)NB * QDIM * CDIM;         // 4 MiB
    float*  uv = (float*)d_out;

    dim3 tb(32, 8);
    dim3 tg(QDIM / 32, CDIM / 32, 2 * NB);
    transpose_cvt_f16<<<tg, tb, 0, stream>>>(x, y, xt, yt);

    dim3 mb(128);
    dim3 mg(QDIM / 64, NB);
    uv_attn_wmma<<<mg, mb, 0, stream>>>(xt, yt, uv);
}
